// GraphAttNet_79542794322138
// MI455X (gfx1250) — matrix operand order repaired and checked by reading, not compiled, not run on hardware
//
#include <hip/hip_runtime.h>
#include <stdint.h>

// ---------------------------------------------------------------- types
typedef __bf16 bf16;
typedef bf16  v8bf  __attribute__((ext_vector_type(8)));
typedef bf16  v16bf __attribute__((ext_vector_type(16)));
typedef float v8f   __attribute__((ext_vector_type(8)));

// ---------------------------------------------------------------- shapes
#define B_SZ   32
#define N_N    36
#define Q_D    1024
#define ENT_D  2048
#define REL_D  512
#define J_D    2560
#define HID_D  512
#define N_ATTH 4
#define OUT_D  1024
#define ROWS   (B_SZ * N_N)          // 1152  (b,s)
#define PAIRS  (ROWS * N_N)          // 41472 (b,s,o)
#define CW_R   1536                  // 1024 head rows + 512 W rows

// ---------------------------------------------------------------- helpers
// A (16x32 bf16, row-major source): lane holds row m = lane%16, 16 contiguous K.
__device__ __forceinline__ v16bf load_a_frag(const bf16* __restrict__ row, int kb) {
    kb += (kb & 8);                            // ks*32 + half*16
    v8bf lo = *(const v8bf*)(row + kb);        // K = kb .. kb+7
    v8bf hi = *(const v8bf*)(row + kb + 8);    // K = kb+8 .. kb+15
    v16bf a;
#pragma unroll
    for (int i = 0; i < 8; ++i) { a[i] = lo[i]; a[i + 8] = hi[i]; }
    return a;
}

// async global -> LDS copy of one 16-byte chunk (CDNA5 ASYNCcnt path)
__device__ __forceinline__ void async_copy16(unsigned lds_off, const bf16* g) {
    asm volatile("global_load_async_to_lds_b128 %0, %1, off"
                 :: "v"(lds_off), "v"(g) : "memory");
}
__device__ __forceinline__ void async_wait0() {
    asm volatile("s_wait_asynccnt 0" ::: "memory");
}

#define WMMA_BF16(a, b, c) \
    __builtin_amdgcn_wmma_f32_16x16x32_bf16(false, (a), false, (b), (short)0, (c), false, false)

// ---------------------------------------------------------------- prep kernels
// CW[row][0:2560] = g * v_row / ||v_row||  (rows 0..1023 head, 1024..1535 W), bf16
__global__ void prep_cw(const float* __restrict__ head_v, const float* __restrict__ head_g,
                        const float* __restrict__ W_v,    const float* __restrict__ W_g,
                        bf16* __restrict__ CW) {
    int row = blockIdx.x, t = threadIdx.x;
    const float* src; float g;
    if (row < 1024) { src = head_v + (size_t)row * J_D;          g = head_g[row]; }
    else            { src = W_v   + (size_t)(row - 1024) * J_D;  g = W_g[row - 1024]; }
    __shared__ float red[256];
    float p = 0.f;
    for (int d = t; d < J_D; d += 256) { float v = src[d]; p += v * v; }
    red[t] = p; __syncthreads();
    for (int s = 128; s > 0; s >>= 1) { if (t < s) red[t] += red[t + s]; __syncthreads(); }
    float scale = g * rsqrtf(red[0]);
    bf16* dst = CW + (size_t)row * J_D;
    for (int d = t; d < J_D; d += 256) dst[d] = (bf16)(src[d] * scale);
}

// Aw[4][512] f32 normalized A weights
__global__ void prep_aw(const float* __restrict__ A_v, const float* __restrict__ A_g,
                        float* __restrict__ Aw) {
    int row = blockIdx.x, t = threadIdx.x;
    const float* src = A_v + (size_t)row * HID_D;
    __shared__ float red[256];
    float p = 0.f;
    for (int d = t; d < HID_D; d += 256) { float v = src[d]; p += v * v; }
    red[t] = p; __syncthreads();
    for (int s = 128; s > 0; s >>= 1) { if (t < s) red[t] += red[t + s]; __syncthreads(); }
    float scale = A_g[row] * rsqrtf(red[0]);
    for (int d = t; d < HID_D; d += 256) Aw[(size_t)row * HID_D + d] = src[d] * scale;
}

// q_h[b][i] = (Q_g[i]/||Q_v[i]||) * dot(q_emb[b], Q_v[i])
__global__ void prep_qh(const float* __restrict__ q_emb, const float* __restrict__ Q_v,
                        const float* __restrict__ Q_g, float* __restrict__ qh) {
    int i = blockIdx.x, t = threadIdx.x;
    const float* vrow = Q_v + (size_t)i * Q_D;
    __shared__ float red[256];
    float p = 0.f;
    for (int d = t; d < Q_D; d += 256) { float v = vrow[d]; p += v * v; }
    red[t] = p; __syncthreads();
    for (int s = 128; s > 0; s >>= 1) { if (t < s) red[t] += red[t + s]; __syncthreads(); }
    float scale = Q_g[i] * rsqrtf(red[0]);
    for (int b = 0; b < B_SZ; ++b) {
        const float* x = q_emb + (size_t)b * Q_D;
        float q = 0.f;
        for (int d = t; d < Q_D; d += 256) q += x[d] * vrow[d];
        __syncthreads();
        red[t] = q; __syncthreads();
        for (int s = 128; s > 0; s >>= 1) { if (t < s) red[t] += red[t + s]; __syncthreads(); }
        if (t == 0) qh[(size_t)b * HID_D + i] = red[0] * scale;
        __syncthreads();
    }
}

__global__ void cvt_bf16(const float* __restrict__ src, bf16* __restrict__ dst, int n) {
    int i = blockIdx.x * blockDim.x + threadIdx.x;
    if (i < n) dst[i] = (bf16)src[i];
}

// ---------------------------------------------------------------- WMMA GEMMs
// E[1152][1536] = ent_bf16[1152][2048] @ CW[:, 0:2048]^T   (4M x 2N register tile)
__global__ void __launch_bounds__(256) gemm_e(const bf16* __restrict__ A,
                                              const bf16* __restrict__ CW,
                                              float* __restrict__ E) {
    const int NMAC = CW_R / 32;                    // 48 macro N-tiles
    int wave = blockIdx.x * 8 + (threadIdx.x >> 5);
    int lane = threadIdx.x & 31;
    int half = lane >> 4;
    int r15  = lane & 15;
    int mt = wave / NMAC, nt = wave % NMAC;
    const bf16* ar[4];
#pragma unroll
    for (int i = 0; i < 4; ++i) ar[i] = A + (size_t)(mt * 64 + i * 16 + r15) * ENT_D;
    const bf16* br[2];
#pragma unroll
    for (int j = 0; j < 2; ++j) br[j] = CW + (size_t)(nt * 32 + j * 16 + r15) * J_D;
    v8f acc[4][2];
#pragma unroll
    for (int i = 0; i < 4; ++i)
#pragma unroll
        for (int j = 0; j < 2; ++j)
#pragma unroll
            for (int r = 0; r < 8; ++r) acc[i][j][r] = 0.f;
    for (int ks = 0; ks < ENT_D / 32; ++ks) {
        int kb = ks * 32 + half * 8;
        __builtin_prefetch(br[0] + (ks + 8) * 32, 0, 1);
        __builtin_prefetch(br[1] + (ks + 8) * 32, 0, 1);
        v16bf a[4], b[2];
#pragma unroll
        for (int i = 0; i < 4; ++i) a[i] = load_a_frag(ar[i], kb);
#pragma unroll
        for (int j = 0; j < 2; ++j) b[j] = *(const v16bf*)(br[j] + ks * 32 + half * 16);
#pragma unroll
        for (int i = 0; i < 4; ++i)
#pragma unroll
            for (int j = 0; j < 2; ++j)
                acc[i][j] = WMMA_BF16(a[i], b[j], acc[i][j]);
    }
#pragma unroll
    for (int i = 0; i < 4; ++i)
#pragma unroll
        for (int j = 0; j < 2; ++j)
#pragma unroll
            for (int r = 0; r < 8; ++r)
                E[(size_t)(mt * 64 + i * 16 + r + half * 8) * CW_R + nt * 32 + j * 16 + r15] =
                    acc[i][j][r];
}

// Wj[41472][512] = rel_bf16 @ CW[1024+n][2048:]^T + E[(m/36)][1024+n]
// A macro-tile (64 x 512 bf16 = 64 KB) staged via async global->LDS, shared by 8 waves.
__global__ void __launch_bounds__(256) gemm_wj(const bf16* __restrict__ A,
                                               const bf16* __restrict__ CW,
                                               const float* __restrict__ E,
                                               float* __restrict__ Wj) {
    __shared__ bf16 smA[64 * REL_D];               // 65536 B
    int t    = threadIdx.x;
    int wave = t >> 5;
    int lane = t & 31;
    int half = lane >> 4;
    int r15  = lane & 15;
    int mt   = blockIdx.x >> 1;                    // 648 macro M rows of 64
    int nt   = ((blockIdx.x & 1) << 3) + wave;     // 16 macro N cols of 32

    {   // stage A: 64 rows x 512 bf16 = 4096 x 16B chunks
        const bf16* src = A + (size_t)mt * 64 * REL_D;
        unsigned base = (unsigned)(uintptr_t)(&smA[0]);
        for (int c = t; c < 4096; c += 256)
            async_copy16(base + (unsigned)c * 16u, src + c * 8);
        async_wait0();
        __syncthreads();
    }

    const bf16* br[2];
#pragma unroll
    for (int j = 0; j < 2; ++j)
        br[j] = CW + (size_t)(1024 + nt * 32 + j * 16 + r15) * J_D + ENT_D;
    v8f acc[4][2];
#pragma unroll
    for (int i = 0; i < 4; ++i)
#pragma unroll
        for (int j = 0; j < 2; ++j)
#pragma unroll
            for (int r = 0; r < 8; ++r) acc[i][j][r] = 0.f;
    for (int ks = 0; ks < REL_D / 32; ++ks) {
        int kb = ks * 32 + half * 8;
        __builtin_prefetch(br[0] + (ks + 8) * 32, 0, 1);
        __builtin_prefetch(br[1] + (ks + 8) * 32, 0, 1);
        v16bf a[4], b[2];
#pragma unroll
        for (int i = 0; i < 4; ++i)
            a[i] = load_a_frag(&smA[(i * 16 + r15) * REL_D], kb);   // ds_load path
#pragma unroll
        for (int j = 0; j < 2; ++j) b[j] = *(const v16bf*)(br[j] + ks * 32 + half * 16);
#pragma unroll
        for (int i = 0; i < 4; ++i)
#pragma unroll
            for (int j = 0; j < 2; ++j)
                acc[i][j] = WMMA_BF16(a[i], b[j], acc[i][j]);
    }
#pragma unroll
    for (int i = 0; i < 4; ++i)
#pragma unroll
        for (int j = 0; j < 2; ++j)
#pragma unroll
            for (int r = 0; r < 8; ++r) {
                int m = mt * 64 + i * 16 + r + half * 8;
                int n = nt * 32 + j * 16 + r15;
                Wj[(size_t)m * HID_D + n] = acc[i][j][r] + E[(size_t)(m / 36) * CW_R + 1024 + n];
            }
}

// ---------------------------------------------------------------- scores + softmax
__global__ void scores_softmax(const float* __restrict__ qh, const float* __restrict__ Wj,
                               const float* __restrict__ Aw, float* __restrict__ att) {
    int bs = blockIdx.x, b = bs / N_N, s = bs % N_N;
    int t  = threadIdx.x;
    __shared__ float red[4][256];
    __shared__ float sc[N_N][N_ATTH];
    const float* qrow = qh + (size_t)b * HID_D;
    for (int o = 0; o < N_N; ++o) {
        const float* w1 = Wj + ((size_t)bs * N_N + o) * HID_D;              // Wj[b,s,o]
        const float* w2 = Wj + ((size_t)(b * N_N + o) * N_N + s) * HID_D;   // Wj[b,o,s]
        float p0 = 0, p1 = 0, p2 = 0, p3 = 0;
        for (int d = t; d < HID_D; d += 256) {
            float x = qrow[d] * w1[d] * w2[d];
            p0 += x * Aw[d];             p1 += x * Aw[HID_D + d];
            p2 += x * Aw[2 * HID_D + d]; p3 += x * Aw[3 * HID_D + d];
        }
        red[0][t] = p0; red[1][t] = p1; red[2][t] = p2; red[3][t] = p3;
        __syncthreads();
        for (int st = 128; st > 0; st >>= 1) {
            if (t < st)
#pragma unroll
                for (int h = 0; h < N_ATTH; ++h) red[h][t] += red[h][t + st];
            __syncthreads();
        }
        if (t < N_ATTH) { float v = red[t][0]; sc[o][t] = v > 0.f ? v : 0.01f * v; }
        __syncthreads();
    }
    if (t < N_ATTH) {
        float mx = -3.0e38f;
        for (int o = 0; o < N_N; ++o) mx = fmaxf(mx, sc[o][t]);
        float ssum = 0.f;
        for (int o = 0; o < N_N; ++o) { float e = __expf(sc[o][t] - mx); sc[o][t] = e; ssum += e; }
        float inv = 1.f / ssum;
        for (int o = 0; o < N_N; ++o)
            att[((size_t)bs * N_N + o) * N_ATTH + t] = sc[o][t] * inv;
    }
}

// ---------------------------------------------------------------- fused output
// out[bs][n] = sum_o att[bs,o,h(n)] * relu(E[bs,n] + rel[bs,o] . CW_head[n,2048:] + bias[n])
__global__ void __launch_bounds__(256) phase_c(const bf16* __restrict__ relb,
                                               const bf16* __restrict__ CW,
                                               const float* __restrict__ E,
                                               const float* __restrict__ head_b,
                                               const float* __restrict__ att,
                                               float* __restrict__ out) {
    __shared__ bf16  smA[48 * REL_D];        // 48 KB (rows 36..47 zero)
    __shared__ float smAtt[48 * N_ATTH];     // att, zero for padded o
    int bs   = blockIdx.x;
    int t    = threadIdx.x;
    int wave = t >> 5;
    int lane = t & 31;
    int half = lane >> 4;
    int orow = lane & 15;

    {   // stage rel rows (36 x 512 bf16 = 2304 x 16B) async; zero-fill padding
        const bf16* src = relb + (size_t)bs * N_N * REL_D;
        unsigned base = (unsigned)(uintptr_t)(&smA[0]);
        for (int c = t; c < (N_N * REL_D) / 8; c += 256)
            async_copy16(base + (unsigned)c * 16u, src + c * 8);
        uint4 z; z.x = z.y = z.z = z.w = 0u;
        uint4* pad = (uint4*)(smA + N_N * REL_D);          // 12 rows = 768 chunks
        for (int c = t; c < 768; c += 256) pad[c] = z;
        if (t < 48 * N_ATTH) {
            int o = t >> 2, h = t & 3;
            smAtt[t] = (o < N_N) ? att[((size_t)bs * N_N + o) * N_ATTH + h] : 0.f;
        }
        async_wait0();
        __syncthreads();
    }

    const float* Erow = E + (size_t)bs * CW_R;
    for (int p = 0; p < 4; ++p) {                         // 2 N-tiles per pass
        int nt0 = wave * 8 + p * 2;
        int n0  = nt0 * 16 + orow;
        int n1  = n0 + 16;
        float c0 = Erow[n0] + head_b[n0];
        float c1 = Erow[n1] + head_b[n1];
        v8f a00, a10, a20, a01, a11, a21;
#pragma unroll
        for (int r = 0; r < 8; ++r) {
            a00[r] = c0; a10[r] = c0; a20[r] = c0;
            a01[r] = c1; a11[r] = c1; a21[r] = c1;
        }
        const bf16* b0row = CW + (size_t)n0 * J_D + ENT_D;
        const bf16* b1row = CW + (size_t)n1 * J_D + ENT_D;
        for (int ks = 0; ks < REL_D / 32; ++ks) {
            int kb = ks * 32 + half * 8;
            v16bf f0 = load_a_frag(&smA[(orow)      * REL_D], kb);
            v16bf f1 = load_a_frag(&smA[(orow + 16) * REL_D], kb);
            v16bf f2 = load_a_frag(&smA[(orow + 32) * REL_D], kb);
            v16bf b0 = *(const v16bf*)(b0row + ks * 32 + half * 16);
            v16bf b1 = *(const v16bf*)(b1row + ks * 32 + half * 16);
            a00 = WMMA_BF16(f0, b0, a00);
            a10 = WMMA_BF16(f1, b0, a10);
            a20 = WMMA_BF16(f2, b0, a20);
            a01 = WMMA_BF16(f0, b1, a01);
            a11 = WMMA_BF16(f1, b1, a11);
            a21 = WMMA_BF16(f2, b1, a21);
        }
        int h0 = n0 >> 8, h1 = n1 >> 8;
        float s0 = 0.f, s1 = 0.f;
#pragma unroll
        for (int r = 0; r < 8; ++r) {
            int oA = r + half * 8;
            s0 += fmaxf(a00[r], 0.f) * smAtt[(oA)      * N_ATTH + h0];
            s0 += fmaxf(a10[r], 0.f) * smAtt[(oA + 16) * N_ATTH + h0];
            s0 += fmaxf(a20[r], 0.f) * smAtt[(oA + 32) * N_ATTH + h0];
            s1 += fmaxf(a01[r], 0.f) * smAtt[(oA)      * N_ATTH + h1];
            s1 += fmaxf(a11[r], 0.f) * smAtt[(oA + 16) * N_ATTH + h1];
            s1 += fmaxf(a21[r], 0.f) * smAtt[(oA + 32) * N_ATTH + h1];
        }
        s0 += __shfl_xor(s0, 16, 32);
        s1 += __shfl_xor(s1, 16, 32);
        if (half == 0) {
            out[(size_t)bs * OUT_D + n0] = s0;
            out[(size_t)bs * OUT_D + n1] = s1;
        }
    }
}

// ---------------------------------------------------------------- launcher
extern "C" void kernel_launch(void* const* d_in, const int* in_sizes, int n_in,
                              void* d_out, int out_size, void* d_ws, size_t ws_size,
                              hipStream_t stream) {
    (void)in_sizes; (void)n_in; (void)out_size; (void)ws_size;
    const float* q_emb  = (const float*)d_in[0];
    const float* ent    = (const float*)d_in[1];
    const float* rel    = (const float*)d_in[2];
    const float* W_v    = (const float*)d_in[3];
    const float* W_g    = (const float*)d_in[4];
    const float* A_v    = (const float*)d_in[5];
    const float* A_g    = (const float*)d_in[6];
    const float* Q_v    = (const float*)d_in[7];
    const float* Q_g    = (const float*)d_in[8];
    const float* head_v = (const float*)d_in[9];
    const float* head_g = (const float*)d_in[10];
    const float* head_b = (const float*)d_in[11];

    float* out     = (float*)d_out;
    float* att_out = out + (size_t)ROWS * OUT_D;     // second tuple element

    char* ws = (char*)d_ws;
    size_t off = 0;
    auto alloc = [&](size_t bytes) {
        void* p = ws + off;
        off = (off + bytes + 255) & ~(size_t)255;
        return p;
    };
    float* qh   = (float*)alloc(sizeof(float) * B_SZ * HID_D);
    float* Aw   = (float*)alloc(sizeof(float) * N_ATTH * HID_D);
    bf16*  CW   = (bf16*) alloc(sizeof(bf16)  * (size_t)CW_R * J_D);
    bf16*  entb = (bf16*) alloc(sizeof(bf16)  * (size_t)ROWS * ENT_D);
    bf16*  relb = (bf16*) alloc(sizeof(bf16)  * (size_t)PAIRS * REL_D);
    float* E    = (float*)alloc(sizeof(float) * (size_t)ROWS * CW_R);
    float* Wj   = (float*)alloc(sizeof(float) * (size_t)PAIRS * HID_D);

    prep_cw<<<CW_R, 256, 0, stream>>>(head_v, head_g, W_v, W_g, CW);
    prep_aw<<<N_ATTH, 256, 0, stream>>>(A_v, A_g, Aw);
    prep_qh<<<HID_D, 256, 0, stream>>>(q_emb, Q_v, Q_g, qh);

    int nEnt = ROWS * ENT_D, nRel = PAIRS * REL_D;
    cvt_bf16<<<(nEnt + 255) / 256, 256, 0, stream>>>(ent, entb, nEnt);
    cvt_bf16<<<(nRel + 255) / 256, 256, 0, stream>>>(rel, relb, nRel);

    // E gemm: (72/4) x (96/2) = 864 wave tiles, 8 waves/block
    gemm_e<<<(18 * 48) / 8, 256, 0, stream>>>(entb, CW, E);
    // Wj gemm: 648 M-macros x 2 N-halves, A staged in LDS per block
    gemm_wj<<<648 * 2, 256, 0, stream>>>(relb, CW, E, Wj);

    scores_softmax<<<ROWS, 256, 0, stream>>>(qh, Wj, Aw, att_out);
    phase_c<<<ROWS, 256, 0, stream>>>(relb, CW, E, head_b, att_out, out);
}
